// ToyNSALlamaLayer_20486994001942
// MI455X (gfx1250) — matrix operand order repaired and checked by reading, not compiled, not run on hardware
//
#include <hip/hip_runtime.h>
#include <hip/hip_bf16.h>
#include <math.h>

// ---------------- problem constants (from reference) ----------------
#define T_SEQ   2048
#define DMODEL  2048
#define IFFN    8192
#define NHQ     16
#define NHKV    4
#define GSZ     4
#define HD      128
#define KERN_W  32
#define STRIDE_W 16
#define BLOCK_SZ 64
#define TOPK    8
#define NBLK    32
#define MWIN    127
#define WINDOW  512
#define ATTN_SCALE 0.08838834764831845f
#define NEG_BIG (-3.4e38f)

typedef __attribute__((ext_vector_type(16))) __bf16 v16bf;
typedef __attribute__((ext_vector_type(8)))  float  v8f;
typedef __attribute__((ext_vector_type(4)))  unsigned int u32x4;
typedef __attribute__((ext_vector_type(8)))  int i32x8;
typedef __attribute__((ext_vector_type(4)))  int i32x4;

__device__ __forceinline__ unsigned short f2bf(float f) {
  unsigned int u = __float_as_uint(f);
  u += 0x7FFFu + ((u >> 16) & 1u);
  return (unsigned short)(u >> 16);
}
__device__ __forceinline__ float bf2f(unsigned short u) {
  return __uint_as_float(((unsigned int)u) << 16);
}

// ---- CDNA5 async global->LDS copy (ASYNCcnt-tracked, ISA 15.18.3 op 98) ----
__device__ __forceinline__ void async_b128(void* lds_ptr, const void* gptr) {
  unsigned lds_addr = (unsigned)(uintptr_t)lds_ptr;
  asm volatile("global_load_async_to_lds_b128 %0, %1, off"
               :: "v"(lds_addr), "v"(gptr) : "memory");
}
__device__ __forceinline__ void async_wait0() {
  asm volatile("s_wait_asynccnt 0x0" ::: "memory");
}

// ---- CDNA5 Tensor Data Mover: 2D bf16 tile (tile_d0 x tile_d1, row stride in elems) ----
// D# packing per ISA 8.3 (group0) / 8.4 (group1); groups 2/3 zero (2D tensor).
#if __has_builtin(__builtin_amdgcn_tensor_load_to_lds)
#define HAVE_TDM 1
#endif

#ifdef HAVE_TDM
__device__ __forceinline__ void tdm_load_2d(void* lds_ptr, const void* gaddr,
                                            unsigned tile_d0, unsigned tile_d1,
                                            unsigned stride_elems) {
  unsigned long long ga = (unsigned long long)(uintptr_t)gaddr;
  u32x4 g0;
  g0[0] = 1u;                                              // count=1, user descriptor
  g0[1] = (unsigned)(uintptr_t)lds_ptr;                    // lds_addr [63:32]
  g0[2] = (unsigned)ga;                                    // global_addr [95:64]
  g0[3] = (unsigned)((ga >> 32) & 0x1FFFFFFu) | (2u << 30);// addr[56:32] | type=2
  i32x8 g1;
  g1[0] = 0x00010000;                                      // data_size=1 (2 bytes)
  g1[1] = (int)((tile_d0 & 0xFFFFu) << 16);                // tensor_dim0 lo16 @ [63:48]
  g1[2] = (int)(((tile_d0 >> 16) & 0xFFFFu) |              // tensor_dim0 hi16
                ((tile_d1 & 0xFFFFu) << 16));              // tensor_dim1 lo16
  g1[3] = (int)(((tile_d1 >> 16) & 0xFFFFu) |              // tensor_dim1 hi16
                ((tile_d0 & 0xFFFFu) << 16));              // tile_dim0 @ [127:112]
  g1[4] = (int)(tile_d1 & 0xFFFFu);                        // tile_dim1 (tile_dim2=0)
  g1[5] = (int)stride_elems;                               // tensor_dim0_stride lo32
  g1[6] = 0;
  g1[7] = 0;
  i32x4 z4 = {0, 0, 0, 0};
#if __has_include(<hip/amd_detail/amd_gfx1250_TDM.h>)
  i32x8 z8 = {0, 0, 0, 0, 0, 0, 0, 0};
  __builtin_amdgcn_tensor_load_to_lds(g0, g1, z4, z4, z8, 0);
#else
  __builtin_amdgcn_tensor_load_to_lds(g0, g1, z4, z4, 0);
#endif
}
__device__ __forceinline__ void tdm_wait0() {
  __builtin_amdgcn_s_wait_tensorcnt((short)0);
}
#endif

// ---------------- RMSNorm: f32 in -> bf16 out ----------------
__global__ void rmsnorm_kernel(const float* __restrict__ x, const float* __restrict__ w,
                               unsigned short* __restrict__ out) {
  int s = blockIdx.x, tid = threadIdx.x;
  __shared__ float red[256];
  const float* row = x + (size_t)s * DMODEL;
  float ss = 0.f;
  for (int i = tid; i < DMODEL; i += 256) { float v = row[i]; ss += v * v; }
  red[tid] = ss; __syncthreads();
  for (int off = 128; off; off >>= 1) { if (tid < off) red[tid] += red[tid + off]; __syncthreads(); }
  float scale = rsqrtf(red[0] / (float)DMODEL + 1e-6f);
  for (int i = tid; i < DMODEL; i += 256)
    out[(size_t)s * DMODEL + i] = f2bf(w[i] * row[i] * scale);
}

// ---------------- weight pack: W[K][N] f32 -> Wt[N][K] bf16 (LDS transpose) ----------------
__global__ void pack_wt_kernel(const float* __restrict__ W, unsigned short* __restrict__ Wt,
                               int K, int N) {
  __shared__ unsigned short lt[64 * 64];
  int kb = blockIdx.y * 64, nb = blockIdx.x * 64;
  int tid = threadIdx.x;
  int row = tid >> 2, c4 = (tid & 3) * 16;
  const float* src = W + (size_t)(kb + row) * N + nb + c4;
  #pragma unroll
  for (int j = 0; j < 16; j += 4) {
    float4 f = *(const float4*)(src + j);
    lt[(c4 + j + 0) * 64 + row] = f2bf(f.x);
    lt[(c4 + j + 1) * 64 + row] = f2bf(f.y);
    lt[(c4 + j + 2) * 64 + row] = f2bf(f.z);
    lt[(c4 + j + 3) * 64 + row] = f2bf(f.w);
  }
  __syncthreads();
  unsigned short* dst = Wt + (size_t)(nb + row) * K + kb + c4;
  *(uint4*)(dst)     = *(const uint4*)&lt[row * 64 + c4];
  *(uint4*)(dst + 8) = *(const uint4*)&lt[row * 64 + c4 + 8];
}

// ---------------- bf16 WMMA GEMM: C[M,N] = A(bf16)[M,K] @ Bt(bf16)[N][K]^T (+resid) ----
// M%128==0, N%128==0, K%32==0. 256 thr, 128x128 tile; wave tile 32x64 = 8 wmma/k-step.
// Both tiles are raw [row][k] bf16 -> staged by TDM (double-buffered) or async fallback.
__global__ void gemm_bf16_wmma(const unsigned short* __restrict__ A,
                               const unsigned short* __restrict__ Bt,
                               const float* __restrict__ resid,
                               float* __restrict__ C, int M, int N, int K) {
  __shared__ unsigned short As[2][128 * 32];
  __shared__ unsigned short Bs[2][128 * 32];
  int tid  = threadIdx.x;
  int lane = tid & 31, wave = tid >> 5;
  int wm = (wave >> 1) * 32;
  int wn = (wave & 1) * 64;
  int gm = blockIdx.y * 128, gn = blockIdx.x * 128;
  int ml = lane & 15, half = lane >> 4;

  const unsigned short* Abase = A  + (size_t)gm * K;
  const unsigned short* Bbase = Bt + (size_t)gn * K;

  v8f acc[2][4];
  #pragma unroll
  for (int i = 0; i < 2; ++i)
    #pragma unroll
    for (int j = 0; j < 4; ++j) acc[i][j] = (v8f){};

#ifdef HAVE_TDM
  if (wave == 0) {
    tdm_load_2d(As[0], Abase, 32, 128, K);
    tdm_load_2d(Bs[0], Bbase, 32, 128, K);
  }
#else
  int srow = tid >> 1, scol = (tid & 1) * 16;
  async_b128(&As[0][srow * 32 + scol], Abase + (size_t)srow * K + scol);
  async_b128(&As[0][srow * 32 + scol + 8], Abase + (size_t)srow * K + scol + 8);
  async_b128(&Bs[0][srow * 32 + scol], Bbase + (size_t)srow * K + scol);
  async_b128(&Bs[0][srow * 32 + scol + 8], Bbase + (size_t)srow * K + scol + 8);
#endif

  int cur = 0;
  for (int kb = 0; kb < K; kb += 32) {
#ifdef HAVE_TDM
    if (wave == 0) tdm_wait0();
#else
    async_wait0();
#endif
    __syncthreads();
    int kn = kb + 32;
    if (kn < K) {
#ifdef HAVE_TDM
      if (wave == 0) {
        tdm_load_2d(As[cur ^ 1], Abase + kn, 32, 128, K);
        tdm_load_2d(Bs[cur ^ 1], Bbase + kn, 32, 128, K);
      }
#else
      async_b128(&As[cur ^ 1][srow * 32 + scol], Abase + (size_t)srow * K + kn + scol);
      async_b128(&As[cur ^ 1][srow * 32 + scol + 8], Abase + (size_t)srow * K + kn + scol + 8);
      async_b128(&Bs[cur ^ 1][srow * 32 + scol], Bbase + (size_t)srow * K + kn + scol);
      async_b128(&Bs[cur ^ 1][srow * 32 + scol + 8], Bbase + (size_t)srow * K + kn + scol + 8);
#endif
    }
    const unsigned int* As32 = (const unsigned int*)As[cur];
    const unsigned int* Bs32 = (const unsigned int*)Bs[cur];
    union bfu { v16bf v; unsigned int u[8]; };
    bfu af[2];
    #pragma unroll
    for (int mf = 0; mf < 2; ++mf) {
      int row = wm + mf * 16 + ml;
      #pragma unroll
      for (int r = 0; r < 4; ++r) af[mf].u[r]     = As32[row * 16 + half * 4 + r];
      #pragma unroll
      for (int r = 0; r < 4; ++r) af[mf].u[4 + r] = As32[row * 16 + 8 + half * 4 + r];
    }
    bfu bfr[4];
    #pragma unroll
    for (int nf = 0; nf < 4; ++nf) {
      int n = wn + nf * 16 + ml;
      #pragma unroll
      for (int r = 0; r < 4; ++r) bfr[nf].u[r]     = Bs32[n * 16 + half * 4 + r];
      #pragma unroll
      for (int r = 0; r < 4; ++r) bfr[nf].u[4 + r] = Bs32[n * 16 + 8 + half * 4 + r];
    }
    #pragma unroll
    for (int mf = 0; mf < 2; ++mf)
      #pragma unroll
      for (int nf = 0; nf < 4; ++nf)
        acc[mf][nf] = __builtin_amdgcn_wmma_f32_16x16x32_bf16(
            false, af[mf].v, false, bfr[nf].v, (short)0, acc[mf][nf], false, false);
    __syncthreads();
    cur ^= 1;
  }
  // epilogue: uniform residual branch (no per-element control flow)
  if (resid != nullptr) {
    #pragma unroll
    for (int mf = 0; mf < 2; ++mf)
      #pragma unroll
      for (int nf = 0; nf < 4; ++nf)
        #pragma unroll
        for (int r = 0; r < 8; ++r) {
          int row = gm + wm + mf * 16 + r + half * 8;
          int col = gn + wn + nf * 16 + ml;
          C[(size_t)row * N + col] = acc[mf][nf][r] + resid[(size_t)row * N + col];
        }
  } else {
    #pragma unroll
    for (int mf = 0; mf < 2; ++mf)
      #pragma unroll
      for (int nf = 0; nf < 4; ++nf)
        #pragma unroll
        for (int r = 0; r < 8; ++r) {
          int row = gm + wm + mf * 16 + r + half * 8;
          int col = gn + wn + nf * 16 + ml;
          C[(size_t)row * N + col] = acc[mf][nf][r];
        }
  }
}

// ---------------- small-N GEMM for Wg (N=48) ----------------
__global__ void gemm_wg_kernel(const unsigned short* __restrict__ xn,
                               const float* __restrict__ Wg, float* __restrict__ gsc) {
  __shared__ float xrow[DMODEL];
  int s = blockIdx.x, tid = threadIdx.x;
  for (int i = tid; i < DMODEL; i += 256) xrow[i] = bf2f(xn[(size_t)s * DMODEL + i]);
  __syncthreads();
  for (int n = tid; n < 3 * NHQ; n += 256) {
    float acc = 0.f;
    for (int kk = 0; kk < DMODEL; ++kk) acc += xrow[kk] * Wg[(size_t)kk * (3 * NHQ) + n];
    gsc[(size_t)s * (3 * NHQ) + n] = acc;
  }
}

// ---------------- RoPE in-place ----------------
__global__ void rope_kernel(float* __restrict__ q, float* __restrict__ k) {
  int s = blockIdx.x, y = blockIdx.y, j = threadIdx.x;
  float* base = (y < NHQ) ? (q + (size_t)s * NHQ * HD + y * HD)
                          : (k + (size_t)s * NHKV * HD + (y - NHQ) * HD);
  float inv = __powf(10000.f, -((float)(2 * j)) / (float)HD);
  float ang = (float)s * inv;
  float c = cosf(ang), sn = sinf(ang);
  float x1 = base[j], x2 = base[j + 64];
  base[j]      = x1 * c - x2 * sn;
  base[j + 64] = x2 * c + x1 * sn;
}

// ---------------- compressed K/V pooling ----------------
__global__ void ckcv_kernel(const float* __restrict__ k, const float* __restrict__ v,
                            float* __restrict__ ck, float* __restrict__ cv) {
  int m = blockIdx.x, h = blockIdx.y, d = threadIdx.x;
  float ak = 0.f, av = 0.f;
  int s0 = m * STRIDE_W;
  for (int i = 0; i < KERN_W; ++i) {
    ak += k[(size_t)(s0 + i) * NHKV * HD + h * HD + d];
    av += v[(size_t)(s0 + i) * NHKV * HD + h * HD + d];
  }
  ck[(size_t)m * NHKV * HD + h * HD + d] = ak * (1.f / KERN_W);
  cv[(size_t)m * NHKV * HD + h * HD + d] = av * (1.f / KERN_W);
}

// ---------------- compressed attention per (s, hq) ----------------
__global__ void cattn_kernel(const float* __restrict__ q, const float* __restrict__ ck,
                             const float* __restrict__ cv, float* __restrict__ cp,
                             float* __restrict__ co) {
  int s = blockIdx.x, hq = blockIdx.y, h = hq >> 2, tid = threadIdx.x;
  __shared__ float qrow[HD];
  __shared__ float prob[128];
  __shared__ float red[128];
  qrow[tid] = q[(size_t)s * NHQ * HD + hq * HD + tid];
  __syncthreads();
  int nvalid = (s >= KERN_W - 1) ? ((s - (KERN_W - 1)) / STRIDE_W + 1) : 0;
  if (nvalid > MWIN) nvalid = MWIN;
  int m = tid;
  float sv = NEG_BIG;
  if (m < nvalid) {
    const float* kr = ck + (size_t)m * NHKV * HD + h * HD;
    float acc = 0.f;
    for (int d = 0; d < HD; ++d) acc += qrow[d] * kr[d];
    sv = acc * ATTN_SCALE;
  }
  red[tid] = sv; __syncthreads();
  for (int off = 64; off; off >>= 1) { if (tid < off) red[tid] = fmaxf(red[tid], red[tid + off]); __syncthreads(); }
  float mx = red[0]; __syncthreads();
  float ev = (m < nvalid) ? __expf(sv - mx) : 0.f;
  red[tid] = ev; __syncthreads();
  for (int off = 64; off; off >>= 1) { if (tid < off) red[tid] += red[tid + off]; __syncthreads(); }
  float sum = red[0];
  float p = (nvalid > 0) ? ev / sum : 0.f;
  if (m < MWIN) cp[((size_t)s * NHQ + hq) * MWIN + m] = p;
  prob[tid] = p; __syncthreads();
  float acc = 0.f;
  for (int mm = 0; mm < nvalid; ++mm)
    acc += prob[mm] * cv[(size_t)mm * NHKV * HD + h * HD + tid];
  co[(size_t)s * NHQ * HD + hq * HD + tid] = acc;
}

// ---------------- block scoring + top-8 per (h, s) ----------------
__global__ void bsel_kernel(const float* __restrict__ cp, unsigned int* __restrict__ selmask) {
  int gid = blockIdx.x * blockDim.x + threadIdx.x;
  if (gid >= NHKV * T_SEQ) return;
  int h = gid / T_SEQ, s = gid % T_SEQ;
  float bs[NBLK];
  int qblk = s >> 6;
  for (int b = 0; b < NBLK; ++b) {
    float acc = 0.f;
    for (int g = 0; g < GSZ; ++g) {
      const float* row = cp + ((size_t)s * NHQ + h * GSZ + g) * MWIN;
      #pragma unroll
      for (int mm = 0; mm < 4; ++mm) {
        int m = 4 * b + mm;
        if (m < MWIN) acc += row[m];
      }
    }
    bool forced = (b < 1) || ((b <= qblk) && (b > qblk - 2));
    bool future = (b * BLOCK_SZ) > s;
    if (forced) acc = 3.4e38f;
    if (future) acc = NEG_BIG;
    bs[b] = acc;
  }
  unsigned int msk = 0;
  for (int it = 0; it < TOPK; ++it) {
    float best = -3.5e38f; int bi = -1;
    for (int b = 0; b < NBLK; ++b)
      if (!((msk >> b) & 1u) && bs[b] > best) { best = bs[b]; bi = b; }
    if (bi >= 0) msk |= 1u << bi;
  }
  selmask[(size_t)h * T_SEQ + s] = msk;
}

// ---------------- selected + sliding-window attention ----------------
__global__ void attn_sw_kernel(const float* __restrict__ q, const float* __restrict__ k,
                               const float* __restrict__ v, const unsigned int* __restrict__ selmask,
                               float* __restrict__ so, float* __restrict__ wo) {
  __shared__ float qs[32 * HD];
  __shared__ float ks[32 * HD];
  __shared__ float vs[32 * HD];
  __shared__ float ps_s[32 * 32];
  __shared__ float ps_w[32 * 32];
  __shared__ float red8[32 * 8];
  __shared__ float rowS[32], rowW[32];

  int tid = threadIdx.x;
  int sq = blockIdx.x * 32, hq = blockIdx.y, h = hq >> 2;
  int srow = tid >> 3, scol = (tid & 7) * 16;
  {
    const float* qg = q + (size_t)(sq + srow) * NHQ * HD + hq * HD + scol;
    float* dst = &qs[srow * HD + scol];
    #pragma unroll
    for (int i = 0; i < 4; ++i) async_b128(dst + 4 * i, qg + 4 * i);
  }
  int irow = tid & 31;
  int jgrp = tid >> 5;
  int si = sq + irow;
  unsigned int smask = selmask[(size_t)h * T_SEQ + si];
  async_wait0();
  __syncthreads();

  const float* kgb = k + (size_t)srow * NHKV * HD + h * HD + scol;
  const float* vgb = v + (size_t)srow * NHKV * HD + h * HD + scol;
  float* kdst = &ks[srow * HD + scol];
  float* vdst = &vs[srow * HD + scol];
  size_t tstep = (size_t)32 * NHKV * HD;

  int tend = sq + 31;
  float lmS = NEG_BIG, lmW = NEG_BIG;
  for (int tb = 0; tb <= tend; tb += 32) {
    const float* kg = kgb + (size_t)(tb >> 5) * tstep;
    #pragma unroll
    for (int i = 0; i < 4; ++i) async_b128(kdst + 4 * i, kg + 4 * i);
    async_wait0();
    __syncthreads();
    for (int idx = 0; idx < 4; ++idx) {
      int j = jgrp + 8 * idx, t = tb + j;
      if (t <= si) {
        float acc = 0.f;
        const float* qp = qs + irow * HD;
        const float* kp = ks + j * HD;
        for (int d = 0; d < HD; ++d) acc += qp[d] * kp[d];
        float scv = acc * ATTN_SCALE;
        if ((smask >> (t >> 6)) & 1u) lmS = fmaxf(lmS, scv);
        if (t > si - WINDOW)          lmW = fmaxf(lmW, scv);
      }
    }
    __syncthreads();
  }
  red8[irow * 8 + jgrp] = lmS; __syncthreads();
  if (tid < 32) { float m = NEG_BIG; for (int g = 0; g < 8; ++g) m = fmaxf(m, red8[tid * 8 + g]); rowS[tid] = m; }
  __syncthreads();
  red8[irow * 8 + jgrp] = lmW; __syncthreads();
  if (tid < 32) { float m = NEG_BIG; for (int g = 0; g < 8; ++g) m = fmaxf(m, red8[tid * 8 + g]); rowW[tid] = m; }
  __syncthreads();
  float mS = rowS[irow], mW = rowW[irow];
  __syncthreads();

  int io = tid >> 3;
  int dbase = (tid & 7) * 16;
  float accS[16], accW[16];
  #pragma unroll
  for (int d2 = 0; d2 < 16; ++d2) { accS[d2] = 0.f; accW[d2] = 0.f; }
  float lsS = 0.f, lsW = 0.f;
  for (int tb = 0; tb <= tend; tb += 32) {
    const float* kg = kgb + (size_t)(tb >> 5) * tstep;
    const float* vg = vgb + (size_t)(tb >> 5) * tstep;
    #pragma unroll
    for (int i = 0; i < 4; ++i) async_b128(kdst + 4 * i, kg + 4 * i);
    #pragma unroll
    for (int i = 0; i < 4; ++i) async_b128(vdst + 4 * i, vg + 4 * i);
    async_wait0();
    __syncthreads();
    for (int idx = 0; idx < 4; ++idx) {
      int j = jgrp + 8 * idx, t = tb + j;
      float eS = 0.f, eW = 0.f;
      if (t <= si) {
        float acc = 0.f;
        const float* qp = qs + irow * HD;
        const float* kp = ks + j * HD;
        for (int d = 0; d < HD; ++d) acc += qp[d] * kp[d];
        float scv = acc * ATTN_SCALE;
        if ((smask >> (t >> 6)) & 1u) eS = __expf(scv - mS);
        if (t > si - WINDOW)          eW = __expf(scv - mW);
      }
      ps_s[irow * 32 + j] = eS; ps_w[irow * 32 + j] = eW;
      lsS += eS; lsW += eW;
    }
    __syncthreads();
    for (int jj = 0; jj < 32; ++jj) {
      float pS = ps_s[io * 32 + jj], pW = ps_w[io * 32 + jj];
      const float* vp = vs + jj * HD + dbase;
      #pragma unroll
      for (int d2 = 0; d2 < 16; ++d2) {
        accS[d2] += pS * vp[d2];
        accW[d2] += pW * vp[d2];
      }
    }
    __syncthreads();
  }
  red8[irow * 8 + jgrp] = lsS; __syncthreads();
  if (tid < 32) { float m = 0.f; for (int g = 0; g < 8; ++g) m += red8[tid * 8 + g]; rowS[tid] = m; }
  __syncthreads();
  red8[irow * 8 + jgrp] = lsW; __syncthreads();
  if (tid < 32) { float m = 0.f; for (int g = 0; g < 8; ++g) m += red8[tid * 8 + g]; rowW[tid] = m; }
  __syncthreads();
  float invS = 1.f / rowS[io];
  float invW = 1.f / rowW[io];
  size_t ob = (size_t)(sq + io) * NHQ * HD + hq * HD + dbase;
  #pragma unroll
  for (int d2 = 0; d2 < 16; ++d2) {
    so[ob + d2] = accS[d2] * invS;
    wo[ob + d2] = accW[d2] * invW;
  }
}

// ---------------- gated merge -> bf16 ----------------
__global__ void merge_kernel(const float* __restrict__ gsc, const float* __restrict__ co,
                             const float* __restrict__ so, const float* __restrict__ wo,
                             unsigned short* __restrict__ attn_bf) {
  int s = blockIdx.x, tid = threadIdx.x;
  for (int i = tid; i < DMODEL; i += 256) {
    int hq = i >> 7;
    float g0 = gsc[(size_t)s * 3 * NHQ + 0 * NHQ + hq];
    float g1 = gsc[(size_t)s * 3 * NHQ + 1 * NHQ + hq];
    float g2 = gsc[(size_t)s * 3 * NHQ + 2 * NHQ + hq];
    g0 = 1.f / (1.f + __expf(-g0));
    g1 = 1.f / (1.f + __expf(-g1));
    g2 = 1.f / (1.f + __expf(-g2));
    size_t idx = (size_t)s * DMODEL + i;
    attn_bf[idx] = f2bf(g0 * co[idx] + g1 * so[idx] + g2 * wo[idx]);
  }
}

// ---------------- SwiGLU elementwise ----------------
__global__ void silu_mul_kernel(const float* __restrict__ gate, const float* __restrict__ up,
                                unsigned short* __restrict__ hb, size_t n) {
  size_t i = (size_t)blockIdx.x * blockDim.x + threadIdx.x;
  if (i < n) {
    float x = gate[i];
    float sg = x / (1.f + __expf(-x));
    hb[i] = f2bf(sg * up[i]);
  }
}

// ---------------- host launcher ----------------
extern "C" void kernel_launch(void* const* d_in, const int* in_sizes, int n_in,
                              void* d_out, int out_size, void* d_ws, size_t ws_size,
                              hipStream_t stream) {
  (void)in_sizes; (void)n_in; (void)out_size; (void)ws_size;
  const float* x     = (const float*)d_in[0];
  const float* anorm = (const float*)d_in[2];
  const float* Wq    = (const float*)d_in[3];
  const float* Wk    = (const float*)d_in[4];
  const float* Wv    = (const float*)d_in[5];
  const float* Wg    = (const float*)d_in[6];
  const float* Wo    = (const float*)d_in[7];
  const float* fnorm = (const float*)d_in[8];
  const float* Wgate = (const float*)d_in[9];
  const float* Wup   = (const float*)d_in[10];
  const float* Wdown = (const float*)d_in[11];
  float* out = (float*)d_out;

  char* base = (char*)d_ws;
  size_t off = 0;
  auto take = [&](size_t bytes) -> void* {
    void* p = base + off;
    off += (bytes + 255) & ~(size_t)255;
    return p;
  };
  unsigned short* xn_bf   = (unsigned short*)take((size_t)T_SEQ * DMODEL * 2);
  float*          qb      = (float*)take((size_t)T_SEQ * NHQ * HD * 4);
  float*          kb      = (float*)take((size_t)T_SEQ * NHKV * HD * 4);
  float*          vb      = (float*)take((size_t)T_SEQ * NHKV * HD * 4);
  float*          gsc     = (float*)take((size_t)T_SEQ * 3 * NHQ * 4);
  float*          ck      = (float*)take((size_t)MWIN * NHKV * HD * 4);
  float*          cv      = (float*)take((size_t)MWIN * NHKV * HD * 4);
  float*          cp      = (float*)take((size_t)T_SEQ * NHQ * MWIN * 4);
  float*          co      = (float*)take((size_t)T_SEQ * NHQ * HD * 4);
  float*          so      = (float*)take((size_t)T_SEQ * NHQ * HD * 4);
  float*          woB     = (float*)take((size_t)T_SEQ * NHQ * HD * 4);
  unsigned int*   selmask = (unsigned int*)take((size_t)NHKV * T_SEQ * 4);
  unsigned short* attn_bf = (unsigned short*)take((size_t)T_SEQ * DMODEL * 2);
  float*          x2      = (float*)take((size_t)T_SEQ * DMODEL * 4);
  unsigned short* h_bf    = (unsigned short*)take((size_t)T_SEQ * DMODEL * 2);
  float*          gateb   = (float*)take((size_t)T_SEQ * IFFN * 4);
  float*          upb     = (float*)take((size_t)T_SEQ * IFFN * 4);
  unsigned short* hb_bf   = (unsigned short*)take((size_t)T_SEQ * IFFN * 2);
  // pre-packed transposed bf16 weights: Wt[N][K]
  unsigned short* wt_q    = (unsigned short*)take((size_t)DMODEL * DMODEL * 2);
  unsigned short* wt_k    = (unsigned short*)take((size_t)DMODEL * (NHKV * HD) * 2);
  unsigned short* wt_v    = (unsigned short*)take((size_t)DMODEL * (NHKV * HD) * 2);
  unsigned short* wt_o    = (unsigned short*)take((size_t)DMODEL * DMODEL * 2);
  unsigned short* wt_gate = (unsigned short*)take((size_t)DMODEL * IFFN * 2);
  unsigned short* wt_up   = (unsigned short*)take((size_t)DMODEL * IFFN * 2);
  unsigned short* wt_down = (unsigned short*)take((size_t)IFFN * DMODEL * 2);

  // ---- pack weights (f32 [K][N] -> bf16 [N][K]) ----
  pack_wt_kernel<<<dim3(DMODEL / 64, DMODEL / 64), 256, 0, stream>>>(Wq, wt_q, DMODEL, DMODEL);
  pack_wt_kernel<<<dim3((NHKV * HD) / 64, DMODEL / 64), 256, 0, stream>>>(Wk, wt_k, DMODEL, NHKV * HD);
  pack_wt_kernel<<<dim3((NHKV * HD) / 64, DMODEL / 64), 256, 0, stream>>>(Wv, wt_v, DMODEL, NHKV * HD);
  pack_wt_kernel<<<dim3(DMODEL / 64, DMODEL / 64), 256, 0, stream>>>(Wo, wt_o, DMODEL, DMODEL);
  pack_wt_kernel<<<dim3(IFFN / 64, DMODEL / 64), 256, 0, stream>>>(Wgate, wt_gate, DMODEL, IFFN);
  pack_wt_kernel<<<dim3(IFFN / 64, DMODEL / 64), 256, 0, stream>>>(Wup, wt_up, DMODEL, IFFN);
  pack_wt_kernel<<<dim3(DMODEL / 64, IFFN / 64), 256, 0, stream>>>(Wdown, wt_down, IFFN, DMODEL);

  rmsnorm_kernel<<<T_SEQ, 256, 0, stream>>>(x, anorm, xn_bf);

  gemm_bf16_wmma<<<dim3(DMODEL / 128, T_SEQ / 128), 256, 0, stream>>>(xn_bf, wt_q, nullptr, qb, T_SEQ, NHQ * HD, DMODEL);
  gemm_bf16_wmma<<<dim3((NHKV * HD) / 128, T_SEQ / 128), 256, 0, stream>>>(xn_bf, wt_k, nullptr, kb, T_SEQ, NHKV * HD, DMODEL);
  gemm_bf16_wmma<<<dim3((NHKV * HD) / 128, T_SEQ / 128), 256, 0, stream>>>(xn_bf, wt_v, nullptr, vb, T_SEQ, NHKV * HD, DMODEL);
  gemm_wg_kernel<<<T_SEQ, 256, 0, stream>>>(xn_bf, Wg, gsc);

  rope_kernel<<<dim3(T_SEQ, NHQ + NHKV), 64, 0, stream>>>(qb, kb);
  ckcv_kernel<<<dim3(MWIN, NHKV), HD, 0, stream>>>(kb, vb, ck, cv);
  cattn_kernel<<<dim3(T_SEQ, NHQ), HD, 0, stream>>>(qb, ck, cv, cp, co);
  bsel_kernel<<<(NHKV * T_SEQ) / 256, 256, 0, stream>>>(cp, selmask);
  attn_sw_kernel<<<dim3(T_SEQ / 32, NHQ), 256, 0, stream>>>(qb, kb, vb, selmask, so, woB);
  merge_kernel<<<T_SEQ, 256, 0, stream>>>(gsc, co, so, woB, attn_bf);

  gemm_bf16_wmma<<<dim3(DMODEL / 128, T_SEQ / 128), 256, 0, stream>>>(attn_bf, wt_o, x, x2, T_SEQ, DMODEL, DMODEL);
  rmsnorm_kernel<<<T_SEQ, 256, 0, stream>>>(x2, fnorm, h_bf);
  gemm_bf16_wmma<<<dim3(IFFN / 128, T_SEQ / 128), 256, 0, stream>>>(h_bf, wt_gate, nullptr, gateb, T_SEQ, IFFN, DMODEL);
  gemm_bf16_wmma<<<dim3(IFFN / 128, T_SEQ / 128), 256, 0, stream>>>(h_bf, wt_up, nullptr, upb, T_SEQ, IFFN, DMODEL);

  size_t nact = (size_t)T_SEQ * IFFN;
  silu_mul_kernel<<<(unsigned)((nact + 255) / 256), 256, 0, stream>>>(gateb, upb, hb_bf, nact);
  gemm_bf16_wmma<<<dim3(DMODEL / 128, T_SEQ / 128), 256, 0, stream>>>(hb_bf, wt_down, x2, out, T_SEQ, DMODEL, IFFN);
}
